// ResidualApproximateBNN_24567212933828
// MI455X (gfx1250) — matrix operand order repaired and checked by reading, not compiled, not run on hardware
//
#include <hip/hip_runtime.h>

// ---------------------------------------------------------------------------
// ResidualApproximateBNN on MI455X (gfx1250).
//  - x converted once f32 -> bf16 ; each layer's W transposed+converted once
//    to WT(N,K) bf16  => every GEMM operand is bf16, tiles are pure copies.
//  - WMMA GEMM: v_wmma_f32_16x16x32_bf16, f32 accumulate, double-buffered LDS,
//    async global->LDS staging (global_load_async_to_lds_b128 / ASYNCcnt),
//    fused per-column activation-select + clip epilogue.
// ---------------------------------------------------------------------------

typedef __attribute__((ext_vector_type(16))) __bf16 v16bf;
typedef __attribute__((ext_vector_type(8)))  __bf16 v8bf;
typedef __attribute__((ext_vector_type(4)))  __bf16 v4bf;
typedef __attribute__((ext_vector_type(8)))  float  v8f;
typedef __attribute__((ext_vector_type(4)))  int    v4i;

#define BM 128
#define BN 128
#define BK 32
#define BKP 40   // padded LDS row stride (bf16): 80B rows -> conflict-free frag reads

#if defined(__HIP_DEVICE_COMPILE__) && __has_builtin(__builtin_amdgcn_global_load_async_to_lds_b128)
#define USE_ASYNC_LDS 1
#else
#define USE_ASYNC_LDS 0
#endif

union BF16Frag { v16bf v; v8bf h[2]; };

__device__ __forceinline__ void wait_async0() {
#if __has_builtin(__builtin_amdgcn_s_wait_asynccnt)
  __builtin_amdgcn_s_wait_asynccnt(0);
#else
  asm volatile("s_wait_asynccnt 0x0" ::: "memory");
#endif
}

__device__ __forceinline__ float act_clip(float v, int id, float mo) {
  float r;
  if (id == 0)      r = fmaxf(v, 0.0f);                       // relu
  else if (id == 1) r = 1.0f / (1.0f + __expf(-v));           // sigmoid
  else if (id == 2) r = tanhf(v);                             // tanh
  else if (id == 3) r = (v >= 0.0f) ? v : 0.1f * v;           // leaky relu 0.1
  else {                                                      // selu
    const float kAlpha = 1.6732632423543772f;
    const float kScale = 1.0507009873554805f;
    r = (v > 0.0f) ? kScale * v : kScale * kAlpha * (__expf(v) - 1.0f);
  }
  return fminf(r, mo);
}

// -------- elementwise f32 -> bf16 (for the x input) ------------------------
__global__ void __launch_bounds__(256)
cvt_f32_bf16(const float* __restrict__ src, __bf16* __restrict__ dst, int n4) {
  const int i = blockIdx.x * 256 + threadIdx.x;
  if (i < n4) {
    float4 f = ((const float4*)src)[i];
    v4bf r; r[0] = (__bf16)f.x; r[1] = (__bf16)f.y;
            r[2] = (__bf16)f.z; r[3] = (__bf16)f.w;
    ((v4bf*)dst)[i] = r;
  }
}

// -------- W(K,N) f32 -> WT(N,K) bf16, LDS-tiled 32x32 transpose ------------
__global__ void __launch_bounds__(256)
transpose_cvt_bf16(const float* __restrict__ W, __bf16* __restrict__ WT,
                   int K, int N) {
  __shared__ __bf16 t[32][33];
  const int tx = threadIdx.x;          // 0..31
  const int ty = threadIdx.y;          // 0..7
  const int k0 = blockIdx.y * 32;
  const int n0 = blockIdx.x * 32;
#pragma unroll
  for (int i = 0; i < 4; ++i) {        // coalesced reads along N
    const int k = k0 + ty + i * 8;
    t[ty + i * 8][tx] = (__bf16)W[(size_t)k * N + n0 + tx];
  }
  __syncthreads();
#pragma unroll
  for (int i = 0; i < 4; ++i) {        // coalesced writes along K
    const int n = n0 + ty + i * 8;
    WT[(size_t)n * K + k0 + tx] = t[tx][ty + i * 8];
  }
}

// C(M,N) = act( [A0 | A1](M,K) @ W(K,N) + bias ), all operands bf16,
// W given as WT(N,K). A0 covers k in [0,K0), A1 covers [K0,K).
// Dims: M%128==0, N%128==0, K%32==0, K0%32==0.
template <int OUT_F32>
__global__ void __launch_bounds__(256)
fused_gemm_act(const __bf16* __restrict__ A0, const __bf16* __restrict__ A1,
               const __bf16* __restrict__ WT, const float* __restrict__ bias,
               const float* __restrict__ mo, const int* __restrict__ aid,
               void* __restrict__ out, int M, int N, int K, int K0)
{
  __shared__ __bf16 sA[2][BM * BKP];   // [m][k]
  __shared__ __bf16 sB[2][BN * BKP];   // [n][k]  (WT tile, already transposed)

  const int tid  = threadIdx.x;
  const int lane = tid & 31;
  const int wid  = tid >> 5;
  const int wm   = wid & 3;    // 0..3 -> 32-row slab of the 128-row block tile
  const int wn   = wid >> 2;   // 0..1 -> 64-col slab of the 128-col block tile

  const int blockRow = blockIdx.y * BM;
  const int blockCol = blockIdx.x * BN;

  const int nk = K / BK;

  // per-lane WMMA fragment addressing (16-bit layouts, ISA 7.12.2)
  const int fragM  = lane & 15;
  const int aKbase = (lane >> 4) << 3;   // A: lanes 0-15 -> K0..7 & K16..23 ; 16-31 -> +8
  const int bKbase = (lane >> 4) << 3;   // B: lanes 0-15 -> K0..7 & K16..23 ; 16-31 -> +8

  v8f acc[2][4];
#pragma unroll
  for (int im = 0; im < 2; ++im)
#pragma unroll
    for (int in = 0; in < 4; ++in)
#pragma unroll
      for (int j = 0; j < 8; ++j) acc[im][in][j] = 0.0f;

  // A-side pointer select per K-tile (concat handled by splitting K at K0)
  auto aSrc = [&](int gk, int row) -> const __bf16* {
    if (gk < K0) return A0 + (size_t)row * K0 + gk;
    return A1 + (size_t)row * (K - K0) + (gk - K0);
  };

  auto compute = [&](int buf) {
    BF16Frag aF[2], bF[4];
#pragma unroll
    for (int im = 0; im < 2; ++im) {
      const __bf16* p = &sA[buf][(wm * 32 + im * 16 + fragM) * BKP + aKbase];
      aF[im].h[0] = *(const v8bf*)p;          // K = kbase+0..7
      aF[im].h[1] = *(const v8bf*)(p + 16);   // K = kbase+16..23
    }
#pragma unroll
    for (int in = 0; in < 4; ++in) {
      const __bf16* p = &sB[buf][(wn * 64 + in * 16 + fragM) * BKP + bKbase];
      bF[in].h[0] = *(const v8bf*)p;          // K = kbase+0..7
      bF[in].h[1] = *(const v8bf*)(p + 16);   // K = kbase+16..23
    }
#pragma unroll
    for (int im = 0; im < 2; ++im)
#pragma unroll
      for (int in = 0; in < 4; ++in)
        acc[im][in] = __builtin_amdgcn_wmma_f32_16x16x32_bf16(
            false, aF[im].v, false, bF[in].v, (short)0, acc[im][in],
            false, false);
  };

#if USE_ASYNC_LDS
  // ---- async global->LDS staging: 4 x b128 per thread per tile ------------
  const int chunk = (tid & 3) * 8;   // 16B chunk within a 32-elem k-row
  const int r0    = tid >> 2;        // 0..63 ; second pass +64
  auto issueAsync = [&](int kt, int buf) {
    const int gk = kt * BK;
#pragma unroll
    for (int t = 0; t < 2; ++t) {
      const int row = r0 + t * 64;
      const __bf16* gA = aSrc(gk, blockRow + row) + chunk;
      const __bf16* gB = WT + (size_t)(blockCol + row) * K + gk + chunk;
      __bf16* lA = &sA[buf][row * BKP + chunk];
      __bf16* lB = &sB[buf][row * BKP + chunk];
      __builtin_amdgcn_global_load_async_to_lds_b128(
          (v4i __attribute__((address_space(1)))*)gA,
          (v4i __attribute__((address_space(3)))*)lA, 0, 0);
      __builtin_amdgcn_global_load_async_to_lds_b128(
          (v4i __attribute__((address_space(1)))*)gB,
          (v4i __attribute__((address_space(3)))*)lB, 0, 0);
    }
  };

  issueAsync(0, 0);
  for (int kt = 0; kt < nk; ++kt) {
    const int cur = kt & 1;
    wait_async0();        // my copies for 'cur' are complete (in-order)
    __syncthreads();      // everyone's copies landed; prior reads of nxt done
    if (kt + 1 < nk) issueAsync(kt + 1, cur ^ 1);  // overlaps the WMMAs below
    compute(cur);
  }
#else
  // ---- fallback: register staging (4 x b64 loads + ds stores per array) ---
  const int sr = tid >> 3;          // row within pass (0..31)
  const int sc = (tid & 7) << 2;    // k within tile  (0,4,..,28)
  v4bf aReg[4], bReg[4];
  auto loadG = [&](int kt) {
    const int gk = kt * BK;
#pragma unroll
    for (int p = 0; p < 4; ++p)
      aReg[p] = *(const v4bf*)(aSrc(gk, blockRow + p * 32 + sr) + sc);
#pragma unroll
    for (int p = 0; p < 4; ++p)
      bReg[p] = *(const v4bf*)(WT + (size_t)(blockCol + p * 32 + sr) * K + gk + sc);
  };
  auto storeL = [&](int buf) {
#pragma unroll
    for (int p = 0; p < 4; ++p)
      *(v4bf*)&sA[buf][(p * 32 + sr) * BKP + sc] = aReg[p];
#pragma unroll
    for (int p = 0; p < 4; ++p)
      *(v4bf*)&sB[buf][(p * 32 + sr) * BKP + sc] = bReg[p];
  };

  loadG(0);
  storeL(0);
  __syncthreads();
  for (int kt = 0; kt < nk; ++kt) {
    const int cur  = kt & 1;
    const bool more = (kt + 1) < nk;
    if (more) loadG(kt + 1);
    compute(cur);
    if (more) storeL(cur ^ 1);
    __syncthreads();
  }
#endif

  // Epilogue: D layout -> VGPR j holds row (base + j or j+8), lane%16 = col.
#pragma unroll
  for (int in = 0; in < 4; ++in) {
    const int col  = blockCol + wn * 64 + in * 16 + fragM;
    const float bcol = bias[col];
    const float mcol = mo[col];
    const int   icol = aid[col];
#pragma unroll
    for (int im = 0; im < 2; ++im) {
      const int rowBase = blockRow + wm * 32 + im * 16 + ((lane >> 4) << 3);
      if (OUT_F32) {
        float* op = (float*)out + (size_t)rowBase * N + col;
#pragma unroll
        for (int j = 0; j < 8; ++j)
          op[(size_t)j * N] = act_clip(acc[im][in][j] + bcol, icol, mcol);
      } else {
        __bf16* op = (__bf16*)out + (size_t)rowBase * N + col;
#pragma unroll
        for (int j = 0; j < 8; ++j)
          op[(size_t)j * N] = (__bf16)act_clip(acc[im][in][j] + bcol, icol, mcol);
      }
    }
  }
}

extern "C" void kernel_launch(void* const* d_in, const int* in_sizes, int n_in,
                              void* d_out, int out_size, void* d_ws, size_t ws_size,
                              hipStream_t stream) {
  (void)in_sizes; (void)n_in; (void)out_size; (void)ws_size;

  constexpr int Bb = 4096, INd = 1024, Xd = 2048, OUTd = 512;

  const float* x      = (const float*)d_in[0];
  const float* W_in   = (const float*)d_in[1];
  const float* b_in   = (const float*)d_in[2];
  const float* Wh[6]  = {(const float*)d_in[3], (const float*)d_in[4],
                         (const float*)d_in[5], (const float*)d_in[6],
                         (const float*)d_in[7], (const float*)d_in[8]};
  const float* bh     = (const float*)d_in[9];
  const float* W_out  = (const float*)d_in[10];
  const float* b_out  = (const float*)d_in[11];
  const float* mo_in  = (const float*)d_in[12];
  const float* mo_h   = (const float*)d_in[13];
  const float* mo_out = (const float*)d_in[14];
  const int*   aid_in = (const int*)d_in[15];
  const int*   aid_h  = (const int*)d_in[16];
  const int*   aid_out= (const int*)d_in[17];

  // workspace: 3 bf16 activation ping-pong buffers + WT buffer + bf16 x  (~76 MB)
  __bf16* ws0  = (__bf16*)d_ws;
  __bf16* ws1  = ws0 + (size_t)Bb * Xd;
  __bf16* ws2  = ws1 + (size_t)Bb * Xd;
  __bf16* wsWT = ws2 + (size_t)Bb * Xd;           // up to N*K = 2048*4096 bf16
  __bf16* wsX  = wsWT + (size_t)2 * Xd * Xd;      // Bb*INd bf16

  const dim3 block(256);
  const dim3 tb(32, 8);
  const dim3 gH(Xd / BN, Bb / BM);     // hidden layers: 16 x 32 blocks
  const dim3 gO(OUTd / BN, Bb / BM);   // output layer:   4 x 32 blocks

  auto xposeW = [&](const float* W, int K, int N) {
    dim3 g(N / 32, K / 32);
    transpose_cvt_bf16<<<g, tb, 0, stream>>>(W, wsWT, K, N);
  };

  // x -> bf16 once
  {
    const int n4 = Bb * INd / 4;
    cvt_f32_bf16<<<n4 / 256, block, 0, stream>>>(x, wsX, n4);
  }

  // h0 = act(x @ W_in + b_in)
  xposeW(W_in, INd, Xd);
  fused_gemm_act<0><<<gH, block, 0, stream>>>(
      wsX, wsX, wsWT, b_in, mo_in, aid_in, ws0, Bb, Xd, INd, INd);
  // outs[1] = act(h0 @ Wh0 + bh0)
  xposeW(Wh[0], Xd, Xd);
  fused_gemm_act<0><<<gH, block, 0, stream>>>(
      ws0, ws0, wsWT, bh + 0 * Xd, mo_h + 0 * Xd, aid_h + 0 * Xd,
      ws1, Bb, Xd, Xd, Xd);
  // outs[2] = act(outs[1] @ Wh1 + bh1)
  xposeW(Wh[1], Xd, Xd);
  fused_gemm_act<0><<<gH, block, 0, stream>>>(
      ws1, ws1, wsWT, bh + 1 * Xd, mo_h + 1 * Xd, aid_h + 1 * Xd,
      ws2, Bb, Xd, Xd, Xd);
  // outs[3] = act(concat(outs[2], outs[1]) @ Wh2 + bh2)   (K = 4096, split)
  xposeW(Wh[2], 2 * Xd, Xd);
  fused_gemm_act<0><<<gH, block, 0, stream>>>(
      ws2, ws1, wsWT, bh + 2 * Xd, mo_h + 2 * Xd, aid_h + 2 * Xd,
      ws0, Bb, Xd, 2 * Xd, Xd);
  // outs[4] = act(outs[3] @ Wh3 + bh3)
  xposeW(Wh[3], Xd, Xd);
  fused_gemm_act<0><<<gH, block, 0, stream>>>(
      ws0, ws0, wsWT, bh + 3 * Xd, mo_h + 3 * Xd, aid_h + 3 * Xd,
      ws1, Bb, Xd, Xd, Xd);
  // outs[5] = act(concat(outs[4], outs[3]) @ Wh4 + bh4)   (K = 4096, split)
  xposeW(Wh[4], 2 * Xd, Xd);
  fused_gemm_act<0><<<gH, block, 0, stream>>>(
      ws1, ws0, wsWT, bh + 4 * Xd, mo_h + 4 * Xd, aid_h + 4 * Xd,
      ws2, Bb, Xd, 2 * Xd, Xd);
  // outs[6] = act(outs[5] @ Wh5 + bh5)
  xposeW(Wh[5], Xd, Xd);
  fused_gemm_act<0><<<gH, block, 0, stream>>>(
      ws2, ws2, wsWT, bh + 5 * Xd, mo_h + 5 * Xd, aid_h + 5 * Xd,
      ws0, Bb, Xd, Xd, Xd);
  // out = act(outs[6] @ W_out + b_out)   (f32 output)
  xposeW(W_out, Xd, OUTd);
  fused_gemm_act<1><<<gO, block, 0, stream>>>(
      ws0, ws0, wsWT, b_out, mo_out, aid_out, d_out, Bb, OUTd, Xd, Xd);
}
